// SessionGraph_3573412790500
// MI455X (gfx1250) — matrix operand order repaired and checked by reading, not compiled, not run on hardware
//
#include <hip/hip_runtime.h>
#include <hip/hip_bf16.h>

// ---------------------------------------------------------------------------
// SGNN (star session-graph GNN) fused kernel for gfx1250 (MI455X, wave32)
//   - one workgroup (256 threads = 8 waves) per batch session
//   - all GEMMs via v_wmma_f32_16x16x32_f16, f32 accumulate
//   - adjacency matrix staged via GLOBAL_LOAD_ASYNC_TO_LDS_B128 (ASYNCcnt),
//     overlapped with the e_in/e_out and input_in WMMA GEMMs
//   - weights pre-converted to f16 into d_ws by a prologue kernel
// Shapes: BS=1024, N=100 (padded 112), H=128, STEP=1
// ---------------------------------------------------------------------------

typedef __attribute__((ext_vector_type(16))) _Float16 v16h;
typedef __attribute__((ext_vector_type(8)))  float    v8f;

#define N_ROW   100
#define N_PAD   112            // 7 tiles of 16
#define HDIM    128
#define H2      256
#define H3      384
#define BSZ     1024
#define STAR_OFF ((size_t)BSZ * N_ROW * HDIM)

// f16 weight workspace layout (element offsets)
#define OFF_EIN  0              // [128 x 128]
#define OFF_EOUT 16384          // [128 x 128]
#define OFF_IH   32768          // [384 x 256]
#define OFF_HH   131072         // [384 x 128]
#define OFF_HN   180224         // [128 x 256]
#define W16_TOTAL 212992

__device__ __forceinline__ float sigmoidf_fast(float x) {
    return 1.0f / (1.0f + __expf(-x));
}

// --- WMMA operand loaders (wave32; A and B share one contiguous k order) -----
// A-matrix 16x32 f16: lane (m = lane&15, half = lane>>4) holds
//   elements 0..15 -> K = koff + half*16 + e, contiguous
__device__ __forceinline__ v16h ldsA16(const _Float16* src, int ld, int r0,
                                       int koff, int lane) {
    int m = lane & 15, half = lane >> 4;
    const _Float16* p = src + (r0 + m) * ld + koff + half * 16;
    v16h a;
#pragma unroll
    for (int e = 0; e < 16; ++e) a[e] = p[e];
    return a;
}

// B-matrix 32x16 f16 with B(k,j) = W[j][koff+k] (row-major W, ld elems/row):
// lane (j = lane&15, half = lane>>4) holds K = koff + half*16 + e, contiguous.
__device__ __forceinline__ v16h gblB16(const _Float16* W, int ld, int j0,
                                       int koff, int lane) {
    int j = lane & 15, half = lane >> 4;
    const _Float16* p = W + (size_t)(j0 + j) * ld + koff + half * 16;
    v16h b;
#pragma unroll
    for (int e = 0; e < 16; ++e) b[e] = p[e];
    return b;
}

// B-matrix from LDS stored row-major as [K x ld] (true column access):
// B(k,j) = src[k*ld + j]
__device__ __forceinline__ v16h ldsB16(const _Float16* src, int ld, int j0,
                                       int koff, int lane) {
    int j = lane & 15, half = lane >> 4;
    v16h b;
#pragma unroll
    for (int e = 0; e < 16; ++e) b[e] = src[(koff + half * 16 + e) * ld + j0 + j];
    return b;
}

__device__ __forceinline__ v8f wmma_f16(v16h a, v16h b, v8f c) {
    return __builtin_amdgcn_wmma_f32_16x16x32_f16(false, a, false, b,
                                                  (short)0, c, false, false);
}

// --- CDNA5 async global -> LDS (ASYNCcnt) ----------------------------------
// 16B per issued op per lane; LDS byte address = low 32 bits of generic ptr.
__device__ __forceinline__ void async_g2l_b128(unsigned lds_addr,
                                               const float* gaddr) {
    asm volatile("global_load_async_to_lds_b128 %0, %1, off"
                 :: "v"(lds_addr), "v"(gaddr) : "memory");
}
__device__ __forceinline__ void wait_asynccnt0() {
    asm volatile("s_wait_asynccnt 0x0" ::: "memory");
}

// ---------------------------------------------------------------------------
// Prologue: convert f32 weights -> f16 workspace (weights then L2-resident)
// ---------------------------------------------------------------------------
__global__ void sgnn_cvt_weights(const float* __restrict__ w_ih,
                                 const float* __restrict__ w_hh,
                                 const float* __restrict__ W_ein,
                                 const float* __restrict__ W_eout,
                                 const float* __restrict__ W_hn,
                                 _Float16* __restrict__ ws) {
    int t = blockIdx.x * blockDim.x + threadIdx.x;
    if (t >= W16_TOTAL) return;
    if (t < OFF_EOUT)       ws[t] = (_Float16)W_ein[t - OFF_EIN];
    else if (t < OFF_IH)    ws[t] = (_Float16)W_eout[t - OFF_EOUT];
    else if (t < OFF_HH)    ws[t] = (_Float16)w_ih[t - OFF_IH];
    else if (t < OFF_HN)    ws[t] = (_Float16)w_hh[t - OFF_HH];
    else                    ws[t] = (_Float16)W_hn[t - OFF_HN];
}

// ---------------------------------------------------------------------------
// Main fused kernel: one block per session.
// ---------------------------------------------------------------------------
__global__ __launch_bounds__(256)
void sgnn_kernel(const int*   __restrict__ inputs,     // [BS, 100]
                 const float* __restrict__ Aadj,       // [BS, 100, 200]
                 const int*   __restrict__ gmask,      // [BS, 100]
                 const float* __restrict__ embedding,  // [N_NODE, 128]
                 const _Float16* __restrict__ w16,     // f16 weights
                 const float* __restrict__ b_ih,       // [384]
                 const float* __restrict__ b_hh,       // [384]
                 const float* __restrict__ b_iah,      // [128]
                 const float* __restrict__ b_oah,      // [128]
                 const float* __restrict__ b_ein,      // [128]
                 const float* __restrict__ b_eout,     // [128]
                 const float* __restrict__ b_hn,       // [128]
                 float* __restrict__ out)              // [BS*100*128 + BS*128]
{
    // ---- LDS (phase-aliased): ~190 KB total ------------------------------
    __shared__ _Float16 sh_h0[N_PAD * HDIM];    // hidden0 f16 (whole kernel)
    __shared__ _Float16 sh_inp[N_PAD * H2];     // inp = [in|out] (C -> D)
    __shared__ _Float16 sh_u[N_PAD * H2];       // B,C: e16 | A16 ; D..F: hnew
    __shared__ __align__(16) float sh_stage[N_PAD * N_PAD]; // async f32 stage
    __shared__ float s_star[HDIM];
    __shared__ float s_row[N_PAD];
    __shared__ float s_gm[N_PAD];
    __shared__ int   s_nid[N_PAD];
    __shared__ float s_cnt;

    _Float16* e16  = sh_u;                   // [N_PAD x 128]
    _Float16* A16  = sh_u + N_PAD * HDIM;    // [N_PAD x 128]
    _Float16* hnew = sh_u;                   // [N_PAD x 128] (from stage D)

    const int b    = blockIdx.x;
    const int tid  = threadIdx.x;
    const int wave = tid >> 5;
    const int lane = tid & 31;
    const int lo   = lane & 15;
    const int hi   = lane >> 4;
    const float inv_sqrt_h = 0.08838834764831845f;   // 1/sqrt(128)

    const _Float16* Wein16  = w16 + OFF_EIN;
    const _Float16* Weout16 = w16 + OFF_EOUT;
    const _Float16* Wih16   = w16 + OFF_IH;
    const _Float16* Whh16   = w16 + OFF_HH;
    const _Float16* Whn16   = w16 + OFF_HN;

    const unsigned stage_base = (unsigned)(size_t)(void*)sh_stage;

    // ============ Stage A: gather hidden0, masked-mean star ===============
    for (int i = tid; i < N_PAD; i += 256) {
        s_nid[i] = (i < N_ROW) ? inputs[b * N_ROW + i] : 0;
        s_gm[i]  = (i < N_ROW) ? (float)gmask[b * N_ROW + i] : 0.0f;
    }
    __syncthreads();

    // kick off async staging of A_in (overlaps gather + e_in GEMM)
    // 100 rows x 400B; 25 b128 chunks per row; 2500 chunks over 256 threads
    for (int q = tid; q < N_ROW * 25; q += 256) {
        int i = q / 25, cq = q % 25;
        const float* g = Aadj + ((size_t)b * N_ROW + i) * (2 * N_ROW) + cq * 4;
        async_g2l_b128(stage_base + (unsigned)((i * N_PAD + cq * 4) * 4), g);
    }

    for (int e = tid; e < N_PAD * HDIM; e += 256) {
        int i = e >> 7, h = e & 127;
        float v = (i < N_ROW) ? embedding[(size_t)s_nid[i] * HDIM + h] : 0.0f;
        sh_h0[e] = (_Float16)v;
    }
    __syncthreads();
    if (tid < HDIM) {
        float acc = 0.0f, cnt = 0.0f;
        for (int i = 0; i < N_ROW; ++i) {
            float g = s_gm[i];
            acc += g * (float)sh_h0[i * HDIM + tid];
            cnt += g;
        }
        s_star[tid] = acc / cnt;
    }
    __syncthreads();

    // ============ Stage B1/C1 and B2/C2: e_{in,out}, adjacency GEMMs ======
    for (int pass = 0; pass < 2; ++pass) {
        const _Float16* We  = pass ? Weout16 : Wein16;
        const float*    be  = pass ? b_eout  : b_ein;
        const float*    bah = pass ? b_oah   : b_iah;
        const int       coff = pass ? HDIM  : 0;   // inp column offset

        // e = hidden0 @ We^T + be     (7x8 = 56 tiles, K=128)
        // (async staging of this pass's adjacency half is in flight here)
        for (int t = wave; t < 56; t += 8) {
            int r = t >> 3, c = t & 7;
            v8f acc = {};
#pragma unroll
            for (int kk = 0; kk < 4; ++kk) {
                v16h a  = ldsA16(sh_h0, HDIM, r * 16, kk * 32, lane);
                v16h bb = gblB16(We, HDIM, c * 16, kk * 32, lane);
                acc = wmma_f16(a, bb, acc);
            }
            float bias = be[c * 16 + lo];
#pragma unroll
            for (int v = 0; v < 8; ++v) {
                int row = r * 16 + v + 8 * hi;
                e16[row * HDIM + c * 16 + lo] = (_Float16)(acc[v] + bias);
            }
        }
        wait_asynccnt0();          // this wave's async staging complete
        __syncthreads();           // all waves' staging visible in LDS

        // staged adjacency f32 -> f16 (zero-padded to 112x128)
        for (int e = tid; e < N_PAD * HDIM; e += 256) {
            int i = e >> 7, j = e & 127;
            float v = (i < N_ROW && j < N_ROW) ? sh_stage[i * N_PAD + j] : 0.0f;
            A16[e] = (_Float16)v;
        }
        __syncthreads();

        // stage buffer free again: prefetch A_out for the next pass,
        // overlapping the input_in GEMM + e_out GEMM below.
        if (pass == 0) {
            for (int q = tid; q < N_ROW * 25; q += 256) {
                int i = q / 25, cq = q % 25;
                const float* g = Aadj + ((size_t)b * N_ROW + i) * (2 * N_ROW)
                                 + N_ROW + cq * 4;
                async_g2l_b128(stage_base + (unsigned)((i * N_PAD + cq * 4) * 4), g);
            }
        }

        // inp_half = A16 @ e16 + bah   (56 tiles, K=128(pad))
        for (int t = wave; t < 56; t += 8) {
            int r = t >> 3, c = t & 7;
            v8f acc = {};
#pragma unroll
            for (int kk = 0; kk < 4; ++kk) {
                v16h a  = ldsA16(A16, HDIM, r * 16, kk * 32, lane);
                v16h bb = ldsB16(e16, HDIM, c * 16, kk * 32, lane);
                acc = wmma_f16(a, bb, acc);
            }
            float bias = bah[c * 16 + lo];
#pragma unroll
            for (int v = 0; v < 8; ++v) {
                int row = r * 16 + v + 8 * hi;
                sh_inp[row * H2 + coff + c * 16 + lo] = (_Float16)(acc[v] + bias);
            }
        }
        __syncthreads();
    }

    // ============ Stage D: gi + gh + fused GRU cell =======================
    // gi = inp @ w_ih^T (+b_ih), gh = hidden0 @ w_hh^T (+b_hh); per 16x16
    // column-tile of H we need the (c), (c+128), (c+256) slices of both.
    // Note: with stride-8 tile assignment, c == wave for all iterations, so
    // B-tile addresses are loop-invariant per wave (CSE-friendly).
    for (int t = wave; t < 56; t += 8) {
        int r = t >> 3, c = t & 7;
        v8f aR = {}, aI = {}, aN = {}, hR = {}, hI = {}, hN = {};
#pragma unroll
        for (int kk = 0; kk < 8; ++kk) {          // gi: K = 256
            v16h a = ldsA16(sh_inp, H2, r * 16, kk * 32, lane);
            aR = wmma_f16(a, gblB16(Wih16, H2, 0   + c * 16, kk * 32, lane), aR);
            aI = wmma_f16(a, gblB16(Wih16, H2, 128 + c * 16, kk * 32, lane), aI);
            aN = wmma_f16(a, gblB16(Wih16, H2, 256 + c * 16, kk * 32, lane), aN);
        }
#pragma unroll
        for (int kk = 0; kk < 4; ++kk) {          // gh: K = 128
            v16h a = ldsA16(sh_h0, HDIM, r * 16, kk * 32, lane);
            hR = wmma_f16(a, gblB16(Whh16, HDIM, 0   + c * 16, kk * 32, lane), hR);
            hI = wmma_f16(a, gblB16(Whh16, HDIM, 128 + c * 16, kk * 32, lane), hI);
            hN = wmma_f16(a, gblB16(Whh16, HDIM, 256 + c * 16, kk * 32, lane), hN);
        }
        int col = c * 16 + lo;
        float biR = b_ih[col], biI = b_ih[HDIM + col], biN = b_ih[H2 + col];
        float bhR = b_hh[col], bhI = b_hh[HDIM + col], bhN = b_hh[H2 + col];
#pragma unroll
        for (int v = 0; v < 8; ++v) {
            int row = r * 16 + v + 8 * hi;
            float rg = sigmoidf_fast((aR[v] + biR) + (hR[v] + bhR));
            float ig = sigmoidf_fast((aI[v] + biI) + (hI[v] + bhI));
            float ng = tanhf((aN[v] + biN) + rg * (hN[v] + bhN));
            float h0 = (float)sh_h0[row * HDIM + col];
            float hv = ng + ig * (h0 - ng);
            hnew[row * HDIM + col] = (_Float16)hv;   // sh_u reused; A16/e16 dead
        }
    }
    __syncthreads();

    // ============ Stage E: star blend + attention pooling =================
    // alpha_i = sigmoid(h_i . star / sqrt(H))
    for (int i = wave; i < N_ROW; i += 8) {
        float p = 0.0f;
        for (int h = lane; h < HDIM; h += 32)
            p += (float)hnew[i * HDIM + h] * s_star[h];
#pragma unroll
        for (int off = 16; off > 0; off >>= 1) p += __shfl_down(p, off);
        if (lane == 0) s_row[i] = sigmoidf_fast(p * inv_sqrt_h);
    }
    __syncthreads();
    for (int e = tid; e < N_ROW * HDIM; e += 256) {
        int i = e >> 7, h = e & 127;
        float a = s_row[i];
        float v = (1.0f - a) * (float)hnew[e] + a * s_star[h];
        hnew[e] = (_Float16)v;
    }
    __syncthreads();
    // attention scores s_i = exp(h_i . star) * gm_i
    for (int i = wave; i < N_ROW; i += 8) {
        float p = 0.0f;
        for (int h = lane; h < HDIM; h += 32)
            p += (float)hnew[i * HDIM + h] * s_star[h];
#pragma unroll
        for (int off = 16; off > 0; off >>= 1) p += __shfl_down(p, off);
        if (lane == 0) s_row[i] = __expf(p) * s_gm[i];
    }
    __syncthreads();
    if (wave == 0) {
        float tsum = 0.0f;
        for (int i = lane; i < N_ROW; i += 32) tsum += s_row[i];
#pragma unroll
        for (int off = 16; off > 0; off >>= 1) tsum += __shfl_down(tsum, off);
        if (lane == 0) s_cnt = tsum + 1e-24f;
    }
    __syncthreads();
    if (tid < HDIM) {
        float acc = 0.0f;
        for (int i = 0; i < N_ROW; ++i)
            acc += s_row[i] * (float)hnew[i * HDIM + tid];
        float st = acc / s_cnt;
        s_star[tid] = st;
        out[STAR_OFF + (size_t)b * HDIM + tid] = st;     // final star output
    }
    __syncthreads();

    // ============ Stage F: highway gate output ============================
    // a = sigmoid([hidden0|hidden] @ W_hn^T + b_hn); out = a*h0 + (1-a)*h
    for (int t = wave; t < 56; t += 8) {
        int r = t >> 3, c = t & 7;
        v8f acc = {};
#pragma unroll
        for (int kk = 0; kk < 8; ++kk) {          // K = 256 (first half h0)
            v16h a = (kk < 4)
                     ? ldsA16(sh_h0, HDIM, r * 16, kk * 32, lane)
                     : ldsA16(hnew,  HDIM, r * 16, (kk - 4) * 32, lane);
            acc = wmma_f16(a, gblB16(Whn16, H2, c * 16, kk * 32, lane), acc);
        }
        int col = c * 16 + lo;
        float bias = b_hn[col];
#pragma unroll
        for (int v = 0; v < 8; ++v) {
            int row = r * 16 + v + 8 * hi;
            if (row < N_ROW) {
                float g  = sigmoidf_fast(acc[v] + bias);
                float h0 = (float)sh_h0[row * HDIM + col];
                float hv = (float)hnew[row * HDIM + col];
                out[((size_t)b * N_ROW + row) * HDIM + col] =
                    g * h0 + (1.0f - g) * hv;
            }
        }
    }
}

// ---------------------------------------------------------------------------
extern "C" void kernel_launch(void* const* d_in, const int* in_sizes, int n_in,
                              void* d_out, int out_size, void* d_ws, size_t ws_size,
                              hipStream_t stream) {
    const int*   inputs  = (const int*)  d_in[0];
    const float* Aadj    = (const float*)d_in[1];
    const int*   gmask   = (const int*)  d_in[2];
    const float* emb     = (const float*)d_in[3];
    const float* w_ih    = (const float*)d_in[4];
    const float* w_hh    = (const float*)d_in[5];
    const float* b_ih    = (const float*)d_in[6];
    const float* b_hh    = (const float*)d_in[7];
    const float* b_iah   = (const float*)d_in[8];
    const float* b_oah   = (const float*)d_in[9];
    const float* W_ein   = (const float*)d_in[10];
    const float* b_ein   = (const float*)d_in[11];
    const float* W_eout  = (const float*)d_in[12];
    const float* b_eout  = (const float*)d_in[13];
    const float* W_hn    = (const float*)d_in[14];
    const float* b_hn    = (const float*)d_in[15];

    _Float16* w16 = (_Float16*)d_ws;

    sgnn_cvt_weights<<<(W16_TOTAL + 255) / 256, 256, 0, stream>>>(
        w_ih, w_hh, W_ein, W_eout, W_hn, w16);

    sgnn_kernel<<<BSZ, 256, 0, stream>>>(
        inputs, Aadj, gmask, emb, w16,
        b_ih, b_hh, b_iah, b_oah, b_ein, b_eout, b_hn,
        (float*)d_out);
}
